// Model_NER_16999480558331
// MI455X (gfx1250) — matrix operand order repaired and checked by reading, not compiled, not run on hardware
//
#include <hip/hip_runtime.h>
#include <hip/hip_bf16.h>

typedef __bf16 bf16;
typedef __attribute__((ext_vector_type(16))) bf16 bf16x16;
typedef __attribute__((ext_vector_type(8))) float f32x8;
typedef __attribute__((ext_vector_type(4))) int v4i;

#if defined(__HIP_DEVICE_COMPILE__) && \
    __has_builtin(__builtin_amdgcn_global_load_async_to_lds_b128) && \
    __has_builtin(__builtin_amdgcn_s_wait_asynccnt)
#define ASYNC_LDS 1
#else
#define ASYNC_LDS 0
#endif

__device__ __forceinline__ bf16 f2bf(float f) {
    unsigned u = __builtin_bit_cast(unsigned, f);
    unsigned r = u + 0x7FFFu + ((u >> 16) & 1u);
    unsigned short h = (unsigned short)(r >> 16);
    return __builtin_bit_cast(bf16, h);
}
__device__ __forceinline__ float sigm(float x) { return 1.f / (1.f + __expf(-x)); }

__device__ __forceinline__ f32x8 wmma_bf16(bf16x16 a, bf16x16 b, f32x8 c) {
#if defined(__HIP_DEVICE_COMPILE__)
    // D = A(16x32 bf16) * B(32x16 bf16) + C(16x16 f32)  -> v_wmma_f32_16x16x32_bf16
    return __builtin_amdgcn_wmma_f32_16x16x32_bf16(false, a, false, b, (short)0, c, false, false);
#else
    (void)a; (void)b; return c;
#endif
}

// 16-byte global -> LDS copy: async (ASYNCcnt-tracked) path if available,
// else VGPR bounce. Builtin params are generic v4i* per compiler diagnostic.
__device__ __forceinline__ void g2l_b128(void* lds, const void* g) {
#if ASYNC_LDS
    __builtin_amdgcn_global_load_async_to_lds_b128((v4i*)g, (v4i*)lds, 0, 0);
#else
    *(uint4*)lds = *(const uint4*)g;
#endif
}
__device__ __forceinline__ void g2l_wait() {
#if ASYNC_LDS
    __builtin_amdgcn_s_wait_asynccnt(0);
#endif
}

// ---------------------------------------------------------------------------
// Generic batched bf16 WMMA GEMM: C[M,N] = A[M,K] @ (B or B^T) (+bias)
// block tile 64x128, 8 waves (4M x 2N), each wave 16x64 (4 C tiles).
// Batch index z -> (bb = z/H, hh = z%H) with independent strides.
// ---------------------------------------------------------------------------
__global__ __launch_bounds__(256) void k_gemm(
    const bf16* __restrict__ A, int lda, long sAb, long sAh,
    const bf16* __restrict__ B, int ldb, long sBb, long sBh, int transB,
    float* Cf, bf16* Cb, int ldc, long sCb, long sCh,
    const float* __restrict__ bias,
    int M, int N, int K, int H)
{
    __shared__ bf16 As[64][32];    // [m][k]
    __shared__ bf16 Bs[128][32];   // [n][k]

    const int tid = threadIdx.x;
    const int wave = tid >> 5, lane = tid & 31;
    const int wm = wave & 3, wn = wave >> 2;
    const int g = lane >> 4, l16 = lane & 15;
    const int m0 = blockIdx.x * 64, n0 = blockIdx.y * 128;
    const int bz = blockIdx.z;
    const int bb = bz / H, hh = bz % H;
    A += (long)bb * sAb + (long)hh * sAh;
    B += (long)bb * sBb + (long)hh * sBh;
    const long coff = (long)bb * sCb + (long)hh * sCh;

    f32x8 acc[4];
    #pragma unroll
    for (int t = 0; t < 4; ++t)
        #pragma unroll
        for (int e = 0; e < 8; ++e) acc[t][e] = 0.f;

    const int aidx = tid * 8, am = aidx >> 5, ak = aidx & 31;

    for (int kk = 0; kk < K; kk += 32) {
        // --- stage A tile (64x32): async global->LDS (b128 per thread) ---
        g2l_b128(&As[am][ak], A + (long)(m0 + am) * lda + kk + ak);
        if (kk + 32 < K)
            __builtin_prefetch(A + (long)(m0 + am) * lda + kk + 32 + ak, 0, 1);
        // --- stage B tile (32x128) as [n][k] ---
        if (!transB) {
            #pragma unroll
            for (int i = 0; i < 16; ++i) {
                int idx = tid + 256 * i;
                int n = idx & 127, k = idx >> 7;
                int gn = n0 + n;
                Bs[n][k] = (gn < N) ? B[(long)(kk + k) * ldb + gn] : f2bf(0.f);
            }
        } else {
            int n = tid >> 1, k16 = (tid & 1) * 16;
            int gn = n0 + n;
            if (gn < N) {
                g2l_b128(&Bs[n][k16], B + (long)gn * ldb + kk + k16);
            } else {
                uint4 z; z.x = z.y = z.z = z.w = 0u;
                *(uint4*)(&Bs[n][k16]) = z;
            }
        }
        g2l_wait();
        __syncthreads();

        // --- A fragment per ISA 16-bit A layout ---
        bf16x16 af;
        #pragma unroll
        for (int j = 0; j < 8; ++j) {
            int kb = ((j < 4) ? 2 * j : 16 + 2 * (j - 4)) + 8 * g;
            af[2 * j]     = As[wm * 16 + l16][kb];
            af[2 * j + 1] = As[wm * 16 + l16][kb + 1];
        }
        // load all B fragments first, then issue WMMAs back-to-back
        bf16x16 bfr[4];
        #pragma unroll
        for (int t = 0; t < 4; ++t) {
            int n = wn * 64 + t * 16 + l16;
            #pragma unroll
            for (int e = 0; e < 16; ++e) bfr[t][e] = Bs[n][((e < 8) ? e : e + 8) + 8 * g];
        }
        #pragma unroll
        for (int t = 0; t < 4; ++t) acc[t] = wmma_bf16(af, bfr[t], acc[t]);
        __syncthreads();
    }

    #pragma unroll
    for (int t = 0; t < 4; ++t) {
        #pragma unroll
        for (int r = 0; r < 8; ++r) {
            int row = m0 + wm * 16 + r + 8 * g;
            int col = n0 + wn * 64 + t * 16 + l16;
            if (col < N) {
                float v = acc[t][r] + (bias ? bias[col] : 0.f);
                long o = coff + (long)row * ldc + col;
                if (Cf) Cf[o] = v;
                if (Cb) Cb[o] = f2bf(v);
            }
        }
    }
}

// ---------------------------------------------------------------------------
// Fused LSTM step: gates = hprev(64x384) @ Wh(384x1536)[gathered 4*32 cols]
//                  + xproj[t] + bias ; pointwise LSTM update.
// grid.x = 12 blocks of 32 hidden units; 8 waves, WMMA recurrent GEMM.
// ---------------------------------------------------------------------------
__global__ __launch_bounds__(256) void k_lstm_step(
    const float* __restrict__ xproj, int tin,
    const bf16* __restrict__ Wh,
    const float* __restrict__ bias,
    const bf16* __restrict__ hprev, bf16* __restrict__ hnext,
    float* __restrict__ cstate,
    bf16* __restrict__ hseq, int colOff, int tout)
{
    __shared__ bf16 As[64][32];
    __shared__ bf16 Bs[128][32];
    __shared__ float Sld[64][128];

    const int tid = threadIdx.x;
    const int wave = tid >> 5, lane = tid & 31;
    const int wm = wave & 3, wn = wave >> 2;
    const int g = lane >> 4, l16 = lane & 15;
    const int u0 = blockIdx.x * 32;

    f32x8 acc[4];
    #pragma unroll
    for (int t = 0; t < 4; ++t)
        #pragma unroll
        for (int e = 0; e < 8; ++e) acc[t][e] = 0.f;

    const int aidx = tid * 8, am = aidx >> 5, ak = aidx & 31;

    for (int kk = 0; kk < 384; kk += 32) {
        g2l_b128(&As[am][ak], hprev + (long)am * 384 + kk + ak);
        #pragma unroll
        for (int i = 0; i < 16; ++i) {
            int idx = tid + 256 * i;
            int c = idx & 127, k = idx >> 7;
            int gate = c >> 5, ui = c & 31;
            Bs[c][k] = Wh[(long)(kk + k) * 1536 + gate * 384 + u0 + ui];
        }
        g2l_wait();
        __syncthreads();

        bf16x16 af;
        #pragma unroll
        for (int j = 0; j < 8; ++j) {
            int kb = ((j < 4) ? 2 * j : 16 + 2 * (j - 4)) + 8 * g;
            af[2 * j]     = As[wm * 16 + l16][kb];
            af[2 * j + 1] = As[wm * 16 + l16][kb + 1];
        }
        bf16x16 bfr[4];
        #pragma unroll
        for (int t = 0; t < 4; ++t) {
            int n = wn * 64 + t * 16 + l16;
            #pragma unroll
            for (int e = 0; e < 16; ++e) bfr[t][e] = Bs[n][((e < 8) ? e : e + 8) + 8 * g];
        }
        #pragma unroll
        for (int t = 0; t < 4; ++t) acc[t] = wmma_bf16(af, bfr[t], acc[t]);
        __syncthreads();
    }

    #pragma unroll
    for (int t = 0; t < 4; ++t)
        #pragma unroll
        for (int r = 0; r < 8; ++r)
            Sld[wm * 16 + r + 8 * g][wn * 64 + t * 16 + l16] = acc[t][r];
    __syncthreads();

    #pragma unroll
    for (int it = 0; it < 8; ++it) {
        int idx = tid + 256 * it;            // 2048 (batch,unit) pairs
        int b = idx >> 5, ui = idx & 31;
        int col = u0 + ui;
        long xrow = ((long)b * 256 + tin) * 1536;
        float iv = Sld[b][ui]      + xproj[xrow + col]         + bias[col];
        float fv = Sld[b][32 + ui] + xproj[xrow + 384 + col]   + bias[384 + col];
        float gv = Sld[b][64 + ui] + xproj[xrow + 768 + col]   + bias[768 + col];
        float ov = Sld[b][96 + ui] + xproj[xrow + 1152 + col]  + bias[1152 + col];
        float co = cstate[b * 384 + col];
        float cn = sigm(fv) * co + sigm(iv) * tanhf(gv);
        float h  = sigm(ov) * tanhf(cn);
        cstate[b * 384 + col] = cn;
        hnext[b * 384 + col] = f2bf(h);
        hseq[((long)b * 256 + tout) * 768 + colOff + col] = f2bf(h);
    }
}

// ---------------------------------------------------------------------------
// Small kernels
// ---------------------------------------------------------------------------
__global__ void k_cvt(const float* __restrict__ src, bf16* __restrict__ dst, long n) {
    long i = (long)blockIdx.x * blockDim.x + threadIdx.x;
    if (i < n) dst[i] = f2bf(src[i]);
}
__global__ void k_zero(unsigned int* p, long n) {
    long i = (long)blockIdx.x * blockDim.x + threadIdx.x;
    if (i < n) p[i] = 0u;
}
__global__ void k_embed(const int* __restrict__ chars, const int* __restrict__ bounds,
                        const int* __restrict__ flags, const int* __restrict__ radicals,
                        const int* __restrict__ pinyins,
                        const float* __restrict__ t_char, const float* __restrict__ t_bound,
                        const float* __restrict__ t_flag, const float* __restrict__ t_rad,
                        const float* __restrict__ t_pin, bf16* __restrict__ out)
{
    int row = blockIdx.x;
    for (int c = threadIdx.x; c < 352; c += blockDim.x) {
        float v;
        if (c < 128)      v = t_char[chars[row] * 128 + c];
        else if (c < 160) v = t_bound[bounds[row] * 32 + (c - 128)];
        else if (c < 224) v = t_flag[flags[row] * 64 + (c - 160)];
        else if (c < 288) v = t_rad[radicals[row] * 64 + (c - 224)];
        else              v = t_pin[pinyins[row] * 64 + (c - 288)];
        out[(long)row * 352 + c] = f2bf(v);
    }
}
__global__ void k_lens(const int* __restrict__ chars, int* __restrict__ lens) {
    __shared__ int red[256];
    int b = blockIdx.x, t = threadIdx.x;
    int c = chars[b * 256 + t];
    red[t] = (c > 0) ? 1 : ((c < 0) ? -1 : 0);
    __syncthreads();
    for (int s = 128; s; s >>= 1) { if (t < s) red[t] += red[t + s]; __syncthreads(); }
    if (t == 0) lens[b] = red[0];
}
__global__ __launch_bounds__(256) void k_softmax(const float* __restrict__ S,
    bf16* __restrict__ P, const int* __restrict__ lens)
{
    int r = blockIdx.x * 8 + (threadIdx.x >> 5);  // row among B*H*T = 65536
    int lane = threadIdx.x & 31;
    int b = r >> 10;
    int len = lens[b];
    const float scale = 0.04419417382415922f;     // 1/sqrt(512)
    float v[8]; float mx = -3.4e38f;
    #pragma unroll
    for (int j = 0; j < 8; ++j) {
        int c = lane + 32 * j;
        float x = S[(long)r * 256 + c];
        x = (c < len) ? x : -4294967296.0f;
        x *= scale;
        v[j] = x; mx = fmaxf(mx, x);
    }
    for (int off = 16; off; off >>= 1) mx = fmaxf(mx, __shfl_xor(mx, off, 32));
    float sum = 0.f;
    #pragma unroll
    for (int j = 0; j < 8; ++j) { v[j] = __expf(v[j] - mx); sum += v[j]; }
    for (int off = 16; off; off >>= 1) sum += __shfl_xor(sum, off, 32);
    float inv = 1.f / sum;
    #pragma unroll
    for (int j = 0; j < 8; ++j) P[(long)r * 256 + lane + 32 * j] = f2bf(v[j] * inv);
}
__global__ void k_crf_score(const float* __restrict__ logits, const int* __restrict__ labels,
                            const int* __restrict__ lens, const float* __restrict__ trans,
                            float* __restrict__ score)
{
    int b = threadIdx.x; if (b >= 64) return;
    int len = lens[b];
    float u = 0.f, bin = 0.f;
    for (int t = 0; t < 256; ++t) {
        int tag = labels[b * 256 + t];
        if (t < len) u += logits[((long)b * 256 + t) * 31 + tag];
        if (t < 255 && t < len - 1) bin += trans[tag * 31 + labels[b * 256 + t + 1]];
    }
    score[b] = u + bin;
}
__global__ void k_crf_lognorm(const float* __restrict__ logits, const int* __restrict__ lens,
                              const float* __restrict__ trans, float* __restrict__ lognorm)
{
    __shared__ float alpha[31], nalpha[31];
    int b = blockIdx.x, j = threadIdx.x;
    int len = lens[b];
    if (j < 31) alpha[j] = logits[((long)b * 256) * 31 + j];
    __syncthreads();
    for (int t = 1; t < len; ++t) {
        if (j < 31) {
            float m = -3.4e38f;
            for (int i = 0; i < 31; ++i) m = fmaxf(m, alpha[i] + trans[i * 31 + j]);
            float s = 0.f;
            for (int i = 0; i < 31; ++i) s += __expf(alpha[i] + trans[i * 31 + j] - m);
            nalpha[j] = logits[((long)b * 256 + t) * 31 + j] + m + __logf(s);
        }
        __syncthreads();
        if (j < 31) alpha[j] = nalpha[j];
        __syncthreads();
    }
    if (j == 0) {
        float m = -3.4e38f;
        for (int i = 0; i < 31; ++i) m = fmaxf(m, alpha[i]);
        float s = 0.f;
        for (int i = 0; i < 31; ++i) s += __expf(alpha[i] - m);
        lognorm[b] = m + __logf(s);
    }
}
__global__ void k_viterbi(const float* __restrict__ logits, const float* __restrict__ trans,
                          float* __restrict__ out)
{
    __shared__ float v[31], nv[31];
    __shared__ unsigned char bp[255][31];
    int b = blockIdx.x, j = threadIdx.x;
    if (j < 31) v[j] = logits[((long)b * 256) * 31 + j];
    __syncthreads();
    for (int t = 1; t < 256; ++t) {
        if (j < 31) {
            float best = -3.4e38f; int bi = 0;
            for (int i = 0; i < 31; ++i) {
                float s = v[i] + trans[i * 31 + j];
                if (s > best) { best = s; bi = i; }
            }
            bp[t - 1][j] = (unsigned char)bi;
            nv[j] = logits[((long)b * 256 + t) * 31 + j] + best;
        }
        __syncthreads();
        if (j < 31) v[j] = nv[j];
        __syncthreads();
    }
    if (j == 0) {
        float best = -3.4e38f; int cur = 0;
        for (int i = 0; i < 31; ++i) if (v[i] > best) { best = v[i]; cur = i; }
        out[1 + (long)b * 256 + 255] = (float)cur;
        for (int t = 254; t >= 0; --t) {
            cur = bp[t][cur];
            out[1 + (long)b * 256 + t] = (float)cur;
        }
    }
}
__global__ void k_loss(const float* __restrict__ score, const float* __restrict__ lognorm,
                       const int* __restrict__ lens, float* __restrict__ out)
{
    __shared__ float red[64];
    int b = threadIdx.x;
    red[b] = score[b] - lognorm[b];
    out[1 + 16384 + b] = (float)lens[b];
    __syncthreads();
    for (int s = 32; s; s >>= 1) { if (b < s) red[b] += red[b + s]; __syncthreads(); }
    if (b == 0) out[0] = -red[0] / 64.f;
}

// ---------------------------------------------------------------------------
extern "C" void kernel_launch(void* const* d_in, const int* in_sizes, int n_in,
                              void* d_out, int out_size, void* d_ws, size_t ws_size,
                              hipStream_t stream)
{
    (void)in_sizes; (void)n_in; (void)out_size; (void)ws_size;
    const int* chars    = (const int*)d_in[0];
    const int* bounds   = (const int*)d_in[1];
    const int* flags    = (const int*)d_in[2];
    const int* radicals = (const int*)d_in[3];
    const int* pinyins  = (const int*)d_in[4];
    const int* labels   = (const int*)d_in[5];
    const float* t_char = (const float*)d_in[6];
    const float* t_bound= (const float*)d_in[7];
    const float* t_flag = (const float*)d_in[8];
    const float* t_rad  = (const float*)d_in[9];
    const float* t_pin  = (const float*)d_in[10];
    const float* Wx1f = (const float*)d_in[11]; const float* Wh1f = (const float*)d_in[12]; const float* b1f = (const float*)d_in[13];
    const float* Wx1b = (const float*)d_in[14]; const float* Wh1b = (const float*)d_in[15]; const float* b1b = (const float*)d_in[16];
    const float* Wx2f = (const float*)d_in[17]; const float* Wh2f = (const float*)d_in[18]; const float* b2f = (const float*)d_in[19];
    const float* Wx2b = (const float*)d_in[20]; const float* Wh2b = (const float*)d_in[21]; const float* b2b = (const float*)d_in[22];
    const float* Wq   = (const float*)d_in[23]; const float* Wk   = (const float*)d_in[24]; const float* Wv = (const float*)d_in[25];
    const float* Wo   = (const float*)d_in[26]; const float* bo   = (const float*)d_in[27];
    const float* Wcrf = (const float*)d_in[28]; const float* trans= (const float*)d_in[29];

    // ---- carve workspace ----
    char* p = (char*)d_ws;
    auto carve = [&](size_t bytes) { void* r = (void*)p; p += (bytes + 255) & ~(size_t)255; return r; };
    bf16* Wx1f_b = (bf16*)carve(352L*1536*2);   bf16* Wx1b_b = (bf16*)carve(352L*1536*2);
    bf16* Wx2f_b = (bf16*)carve(768L*1536*2);   bf16* Wx2b_b = (bf16*)carve(768L*1536*2);
    bf16* Wh1f_b = (bf16*)carve(384L*1536*2);   bf16* Wh1b_b = (bf16*)carve(384L*1536*2);
    bf16* Wh2f_b = (bf16*)carve(384L*1536*2);   bf16* Wh2b_b = (bf16*)carve(384L*1536*2);
    bf16* Wq_b   = (bf16*)carve(768L*512*2);    bf16* Wk_b   = (bf16*)carve(768L*512*2);
    bf16* Wv_b   = (bf16*)carve(768L*512*2);    bf16* Wo_b   = (bf16*)carve(512L*512*2);
    bf16* Wcrf_b = (bf16*)carve(512L*31*2);
    bf16* emb_b  = (bf16*)carve(16384L*352*2);
    float* xproj = (float*)carve(16384L*1536*4);
    bf16* h1seq  = (bf16*)carve(16384L*768*2);
    bf16* h2seq  = (bf16*)carve(16384L*768*2);
    bf16* hstate = (bf16*)carve(2L*64*384*2);
    float* cstate= (float*)carve(64L*384*4);
    bf16* Qb     = (bf16*)carve(16384L*512*2);
    bf16* Kb     = (bf16*)carve(16384L*512*2);
    bf16* Vb     = (bf16*)carve(16384L*512*2);
    float* Sf    = (float*)carve(64L*4*256*256*4);
    bf16* Pb     = (bf16*)carve(64L*4*256*256*2);
    bf16* Zb     = (bf16*)carve(16384L*512*2);
    bf16* AOb    = (bf16*)carve(16384L*512*2);
    float* logitsf = (float*)carve(16384L*31*4);
    int*   lensb   = (int*)carve(64*4);
    float* scoreb  = (float*)carve(64*4);
    float* lognormb= (float*)carve(64*4);

    auto cvt = [&](const float* s, bf16* d, long n) {
        k_cvt<<<dim3((unsigned)((n + 255) / 256)), 256, 0, stream>>>(s, d, n);
    };
    cvt(Wx1f, Wx1f_b, 352L*1536);  cvt(Wx1b, Wx1b_b, 352L*1536);
    cvt(Wx2f, Wx2f_b, 768L*1536);  cvt(Wx2b, Wx2b_b, 768L*1536);
    cvt(Wh1f, Wh1f_b, 384L*1536);  cvt(Wh1b, Wh1b_b, 384L*1536);
    cvt(Wh2f, Wh2f_b, 384L*1536);  cvt(Wh2b, Wh2b_b, 384L*1536);
    cvt(Wq, Wq_b, 768L*512); cvt(Wk, Wk_b, 768L*512); cvt(Wv, Wv_b, 768L*512);
    cvt(Wo, Wo_b, 512L*512); cvt(Wcrf, Wcrf_b, 512L*31);

    k_lens<<<64, 256, 0, stream>>>(chars, lensb);
    k_embed<<<16384, 128, 0, stream>>>(chars, bounds, flags, radicals, pinyins,
                                       t_char, t_bound, t_flag, t_rad, t_pin, emb_b);

    auto run_lstm = [&](const bf16* xin, int K, const bf16* Wx_b, const bf16* Wh_b,
                        const float* bias, bf16* hseq, int colOff, bool rev) {
        // xproj = xin @ Wx  (M=16384, N=1536)
        k_gemm<<<dim3(256, 12, 1), 256, 0, stream>>>(
            xin, K, 0, 0, Wx_b, 1536, 0, 0, 0,
            xproj, nullptr, 1536, 0, 0, nullptr, 16384, 1536, K, 1);
        k_zero<<<96, 256, 0, stream>>>((unsigned*)hstate, 24576);
        k_zero<<<96, 256, 0, stream>>>((unsigned*)cstate, 24576);
        for (int s = 0; s < 256; ++s) {
            int tin = rev ? 255 - s : s;
            const bf16* hp = hstate + (size_t)(s & 1) * (64 * 384);
            bf16* hn = hstate + (size_t)((s + 1) & 1) * (64 * 384);
            k_lstm_step<<<12, 256, 0, stream>>>(xproj, tin, Wh_b, bias, hp, hn,
                                                cstate, hseq, colOff, s);
        }
    };
    run_lstm(emb_b, 352, Wx1f_b, Wh1f_b, b1f, h1seq, 0,   false);
    run_lstm(emb_b, 352, Wx1b_b, Wh1b_b, b1b, h1seq, 384, true);
    run_lstm(h1seq, 768, Wx2f_b, Wh2f_b, b2f, h2seq, 0,   false);
    run_lstm(h1seq, 768, Wx2b_b, Wh2b_b, b2b, h2seq, 384, true);

    // Q, K, V projections (bf16 out)
    k_gemm<<<dim3(256, 4, 1), 256, 0, stream>>>(h2seq, 768, 0, 0, Wq_b, 512, 0, 0, 0,
        nullptr, Qb, 512, 0, 0, nullptr, 16384, 512, 768, 1);
    k_gemm<<<dim3(256, 4, 1), 256, 0, stream>>>(h2seq, 768, 0, 0, Wk_b, 512, 0, 0, 0,
        nullptr, Kb, 512, 0, 0, nullptr, 16384, 512, 768, 1);
    k_gemm<<<dim3(256, 4, 1), 256, 0, stream>>>(h2seq, 768, 0, 0, Wv_b, 512, 0, 0, 0,
        nullptr, Vb, 512, 0, 0, nullptr, 16384, 512, 768, 1);

    // scores S[b,h] = Q_h @ K_h^T  (batched over 256 (b,h) pairs)
    k_gemm<<<dim3(4, 2, 256), 256, 0, stream>>>(
        Qb, 512, 131072L, 128L, Kb, 512, 131072L, 128L, 1,
        Sf, nullptr, 256, 262144L, 65536L, nullptr, 256, 256, 128, 4);

    k_softmax<<<8192, 256, 0, stream>>>(Sf, Pb, lensb);

    // Z[b,h] = P @ V_h  -> scattered into Zb[row][h*128..]
    k_gemm<<<dim3(4, 1, 256), 256, 0, stream>>>(
        Pb, 256, 262144L, 65536L, Vb, 512, 131072L, 128L, 0,
        nullptr, Zb, 512, 131072L, 128L, nullptr, 256, 128, 256, 4);

    // attention output projection (+bo)
    k_gemm<<<dim3(256, 4, 1), 256, 0, stream>>>(Zb, 512, 0, 0, Wo_b, 512, 0, 0, 0,
        nullptr, AOb, 512, 0, 0, bo, 16384, 512, 512, 1);

    // logits = attn_out @ Wcrf (N=31, edge-guarded)
    k_gemm<<<dim3(256, 1, 1), 256, 0, stream>>>(AOb, 512, 0, 0, Wcrf_b, 31, 0, 0, 0,
        logitsf, nullptr, 31, 0, 0, nullptr, 16384, 31, 512, 1);

    // CRF
    k_crf_score<<<1, 64, 0, stream>>>(logitsf, labels, lensb, trans, scoreb);
    k_crf_lognorm<<<64, 32, 0, stream>>>(logitsf, lensb, trans, lognormb);
    k_viterbi<<<64, 32, 0, stream>>>(logitsf, trans, (float*)d_out);
    k_loss<<<1, 64, 0, stream>>>(scoreb, lognormb, lensb, (float*)d_out);
}
